// RNAMPNN_9354438771109
// MI455X (gfx1250) — compile-verified, matrix operand order checked by reading, not run on hardware
//
#include <hip/hip_runtime.h>
#include <hip/hip_bf16.h>

typedef __attribute__((ext_vector_type(16))) __bf16 v16bf;
typedef __attribute__((ext_vector_type(8)))  float  v8f;

#define HN 128
#define NN 16384
#define EE 491520L
#define BB 32
#define VV 4

struct Pack32 { uint4 lo; uint4 hi; };

__device__ __forceinline__ __bf16 f2bf(float x) { return (__bf16)x; }

__device__ __forceinline__ unsigned pack2(float a, float b) {
    unsigned short ua = __builtin_bit_cast(unsigned short, (__bf16)a);
    unsigned short ub = __builtin_bit_cast(unsigned short, (__bf16)b);
    return (unsigned)ua | ((unsigned)ub << 16);
}

// A-fragment (16x32 bf16) for one lane: row fixed, K = k0 + half*8 + {0..7} and +16.
__device__ __forceinline__ v16bf frag_a(const __bf16* rowbase, int k0, int half) {
    Pack32 t;
    t.lo = *reinterpret_cast<const uint4*>(rowbase + k0 + half * 8);
    t.hi = *reinterpret_cast<const uint4*>(rowbase + k0 + 16 + half * 8);
    return __builtin_bit_cast(v16bf, t);
}

// B-fragment from pre-swizzled weights: 32 contiguous bytes per lane.
__device__ __forceinline__ v16bf frag_b(const __bf16* p) {
    const uint4* q = reinterpret_cast<const uint4*>(p);
    Pack32 t; t.lo = q[0]; t.hi = q[1];
    return __builtin_bit_cast(v16bf, t);
}

__device__ __forceinline__ v8f wmma_bf16(v16bf a, v16bf b, v8f c) {
    return __builtin_amdgcn_wmma_f32_16x16x32_bf16(false, a, false, b, (short)0, c, false, false);
}

__device__ __forceinline__ v8f vzero8() { return (v8f){0.f,0.f,0.f,0.f,0.f,0.f,0.f,0.f}; }

// ---------- prep / elementwise kernels ----------

// Swizzle f32 weight [K,N] row-major into per-fragment bf16 layout.
__global__ void swz_kernel(const float* __restrict__ W, __bf16* __restrict__ out, int K, int N) {
    int g = blockIdx.x * blockDim.x + threadIdx.x;
    int NT = N >> 4;
    int frag = g >> 9;
    int lane = (g >> 4) & 31;
    int e = g & 15;
    int v = e >> 1, j = e & 1;
    int half = lane >> 4, c = lane & 15;
    int kt = frag / NT, nt = frag % NT;
    int k = kt * 32 + ((v & 4) ? 16 : 0) + half * 8 + (v & 3) * 2 + j;
    int n = nt * 16 + c;
    out[g] = f2bf(W[(size_t)k * N + n]);
}

__global__ void copyf_kernel(float* __restrict__ dst, const float* __restrict__ src, int n) {
    int i = blockIdx.x * blockDim.x + threadIdx.x;
    if (i < n) dst[i] = src[i];
}

__global__ void zerof_kernel(float* __restrict__ dst, int n) {
    int i = blockIdx.x * blockDim.x + threadIdx.x;
    if (i < n) dst[i] = 0.f;
}

__global__ void cvtbf_kernel(__bf16* __restrict__ dst, const float* __restrict__ src, int n) {
    int i = blockIdx.x * blockDim.x + threadIdx.x;
    if (i < n) dst[i] = f2bf(src[i]);
}

__global__ void copyi_kernel(int* __restrict__ dst, const int* __restrict__ src, int n) {
    int i = blockIdx.x * blockDim.x + threadIdx.x;
    if (i < n) dst[i] = src[i];
}

// ---------- edge MLP + scatter (WMMA) ----------
// 64 edges per block, 128 threads (4 waves). Each wave owns 16 rows.
__global__ __launch_bounds__(128) void edge_kernel(
    const float* __restrict__ hE, const __bf16* __restrict__ hvb,
    const int* __restrict__ Eidx, long Etot,
    const __bf16* __restrict__ w1, const __bf16* __restrict__ w2, const __bf16* __restrict__ w3,
    const float* __restrict__ b1, const float* __restrict__ b2, const float* __restrict__ b3,
    float* __restrict__ dh)
{
    __shared__ alignas(16) __bf16 Xs[64 * 384];   // concat(h_E, h_V[src], h_V[dst]) bf16
    __shared__ alignas(16) __bf16 M1[64 * 128];
    __shared__ alignas(16) __bf16 M2[64 * 128];
    __shared__ int srcS[64];

    int tid = threadIdx.x;
    size_t e0 = (size_t)blockIdx.x * 64;

    if (tid < 64) srcS[tid] = Eidx[e0 + tid];

    { // stage inputs: 2 threads per row
        int r = tid >> 1, hh = tid & 1;
        size_t e = e0 + r;
        int de = Eidx[(size_t)Etot + e];
        int se = Eidx[e];
        const float4* hp = reinterpret_cast<const float4*>(hE + e * HN) + hh * 16;
        __bf16* xr = Xs + r * 384;
        #pragma unroll
        for (int i = 0; i < 16; ++i) {
            float4 f = hp[i];
            uint2 u; u.x = pack2(f.x, f.y); u.y = pack2(f.z, f.w);
            *reinterpret_cast<uint2*>(xr + hh * 64 + i * 4) = u;
        }
        const uint4* sp = reinterpret_cast<const uint4*>(hvb + (size_t)se * HN) + hh * 8;
        uint4* ds = reinterpret_cast<uint4*>(xr + 128) + hh * 8;
        #pragma unroll
        for (int i = 0; i < 8; ++i) ds[i] = sp[i];
        const uint4* dp = reinterpret_cast<const uint4*>(hvb + (size_t)de * HN) + hh * 8;
        uint4* dd = reinterpret_cast<uint4*>(xr + 256) + hh * 8;
        #pragma unroll
        for (int i = 0; i < 8; ++i) dd[i] = dp[i];
    }
    __syncthreads();

    int lane = tid & 31, wv = tid >> 5;
    int half = lane >> 4, c = lane & 15;

    // ---- stage 1: m1 = relu(X @ W1 + b1), K=384 ----
    {
        const __bf16* Arow = Xs + ((wv << 4) + c) * 384;
        v8f acc[8];
        #pragma unroll
        for (int i = 0; i < 8; ++i) acc[i] = vzero8();
        for (int kt = 0; kt < 12; ++kt) {
            v16bf a = frag_a(Arow, kt * 32, half);
            #pragma unroll
            for (int nt = 0; nt < 8; ++nt) {
                v16bf b = frag_b(w1 + ((size_t)(kt * 8 + nt) * 512) + lane * 16);
                acc[nt] = wmma_bf16(a, b, acc[nt]);
            }
        }
        #pragma unroll
        for (int nt = 0; nt < 8; ++nt) {
            float bias = b1[nt * 16 + c];
            #pragma unroll
            for (int v = 0; v < 8; ++v) {
                int rr = (wv << 4) + (half << 3) + v;
                M1[rr * 128 + nt * 16 + c] = f2bf(fmaxf(acc[nt][v] + bias, 0.f));
            }
        }
    }
    __syncthreads();

    // ---- stage 2: m2 = relu(m1 @ W2 + b2), K=128 ----
    {
        const __bf16* Arow = M1 + ((wv << 4) + c) * 128;
        v8f acc[8];
        #pragma unroll
        for (int i = 0; i < 8; ++i) acc[i] = vzero8();
        for (int kt = 0; kt < 4; ++kt) {
            v16bf a = frag_a(Arow, kt * 32, half);
            #pragma unroll
            for (int nt = 0; nt < 8; ++nt) {
                v16bf b = frag_b(w2 + ((size_t)(kt * 8 + nt) * 512) + lane * 16);
                acc[nt] = wmma_bf16(a, b, acc[nt]);
            }
        }
        #pragma unroll
        for (int nt = 0; nt < 8; ++nt) {
            float bias = b2[nt * 16 + c];
            #pragma unroll
            for (int v = 0; v < 8; ++v) {
                int rr = (wv << 4) + (half << 3) + v;
                M2[rr * 128 + nt * 16 + c] = f2bf(fmaxf(acc[nt][v] + bias, 0.f));
            }
        }
    }
    __syncthreads();

    // ---- stage 3: m3 = m2 @ W3 + b3; scatter-add m3/SCALE into dh[src] ----
    {
        const __bf16* Arow = M2 + ((wv << 4) + c) * 128;
        v8f acc[8];
        #pragma unroll
        for (int i = 0; i < 8; ++i) acc[i] = vzero8();
        for (int kt = 0; kt < 4; ++kt) {
            v16bf a = frag_a(Arow, kt * 32, half);
            #pragma unroll
            for (int nt = 0; nt < 8; ++nt) {
                v16bf b = frag_b(w3 + ((size_t)(kt * 8 + nt) * 512) + lane * 16);
                acc[nt] = wmma_bf16(a, b, acc[nt]);
            }
        }
        const float inv_scale = 1.0f / 30.0f;
        #pragma unroll
        for (int nt = 0; nt < 8; ++nt) {
            float bias = b3[nt * 16 + c];
            #pragma unroll
            for (int v = 0; v < 8; ++v) {
                int rr = (wv << 4) + (half << 3) + v;
                int s = srcS[rr];
                atomicAdd(dh + (size_t)s * HN + nt * 16 + c, (acc[nt][v] + bias) * inv_scale);
            }
        }
    }
}

// ---------- node feed-forward 128->512->128 (WMMA) ----------
__global__ __launch_bounds__(128) void ff_kernel(
    const float* __restrict__ hv,
    const __bf16* __restrict__ wd1, const __bf16* __restrict__ wd2,
    const float* __restrict__ bd1, const float* __restrict__ bd2,
    float* __restrict__ out)
{
    __shared__ alignas(16) __bf16 Xs[64 * 128];
    __shared__ alignas(16) __bf16 M1[64 * 512];

    int tid = threadIdx.x;
    size_t row0 = (size_t)blockIdx.x * 64;

    { // load + convert h_V rows
        int r = tid >> 1, hh = tid & 1;
        const float4* p = reinterpret_cast<const float4*>(hv + (row0 + r) * HN) + hh * 16;
        __bf16* xr = Xs + r * 128 + hh * 64;
        #pragma unroll
        for (int i = 0; i < 16; ++i) {
            float4 f = p[i];
            uint2 u; u.x = pack2(f.x, f.y); u.y = pack2(f.z, f.w);
            *reinterpret_cast<uint2*>(xr + i * 4) = u;
        }
    }
    __syncthreads();

    int lane = tid & 31, wv = tid >> 5;
    int half = lane >> 4, c = lane & 15;

    // ---- stage 1: m1 = relu(X @ Wd1 + bd1), [64,128]@[128,512] ----
    {
        const __bf16* Arow = Xs + ((wv << 4) + c) * 128;
        v16bf a4[4];
        #pragma unroll
        for (int kt = 0; kt < 4; ++kt) a4[kt] = frag_a(Arow, kt * 32, half);
        for (int nc = 0; nc < 4; ++nc) {
            v8f acc[8];
            #pragma unroll
            for (int i = 0; i < 8; ++i) acc[i] = vzero8();
            #pragma unroll
            for (int kt = 0; kt < 4; ++kt) {
                #pragma unroll
                for (int q = 0; q < 8; ++q) {
                    int nt = nc * 8 + q;
                    v16bf b = frag_b(wd1 + ((size_t)(kt * 32 + nt) * 512) + lane * 16);
                    acc[q] = wmma_bf16(a4[kt], b, acc[q]);
                }
            }
            #pragma unroll
            for (int q = 0; q < 8; ++q) {
                int nt = nc * 8 + q;
                float bias = bd1[nt * 16 + c];
                #pragma unroll
                for (int v = 0; v < 8; ++v) {
                    int rr = (wv << 4) + (half << 3) + v;
                    M1[rr * 512 + nt * 16 + c] = f2bf(fmaxf(acc[q][v] + bias, 0.f));
                }
            }
        }
    }
    __syncthreads();

    // ---- stage 2: ff = m1 @ Wd2 + bd2, [64,512]@[512,128] ----
    {
        const __bf16* Arow = M1 + ((wv << 4) + c) * 512;
        v8f acc[8];
        #pragma unroll
        for (int i = 0; i < 8; ++i) acc[i] = vzero8();
        for (int kt = 0; kt < 16; ++kt) {
            v16bf a = frag_a(Arow, kt * 32, half);
            #pragma unroll
            for (int nt = 0; nt < 8; ++nt) {
                v16bf b = frag_b(wd2 + ((size_t)(kt * 8 + nt) * 512) + lane * 16);
                acc[nt] = wmma_bf16(a, b, acc[nt]);
            }
        }
        #pragma unroll
        for (int nt = 0; nt < 8; ++nt) {
            float bias = bd2[nt * 16 + c];
            #pragma unroll
            for (int v = 0; v < 8; ++v) {
                int rr = (wv << 4) + (half << 3) + v;
                out[(row0 + rr) * HN + nt * 16 + c] = acc[nt][v] + bias;
            }
        }
    }
}

// ---------- LayerNorm(x + res) * g + b, one row per block ----------
__global__ __launch_bounds__(128) void ln_kernel(
    const float* __restrict__ x, const float* __restrict__ res,
    const float* __restrict__ g, const float* __restrict__ b,
    float* __restrict__ out)
{
    __shared__ float red[128];
    int row = blockIdx.x, t = threadIdx.x;
    float v = x[(size_t)row * HN + t] + res[(size_t)row * HN + t];
    red[t] = v;
    __syncthreads();
    for (int s = 64; s > 0; s >>= 1) { if (t < s) red[t] += red[t + s]; __syncthreads(); }
    float mean = red[0] * (1.0f / HN);
    __syncthreads();
    float d = v - mean;
    red[t] = d * d;
    __syncthreads();
    for (int s = 64; s > 0; s >>= 1) { if (t < s) red[t] += red[t + s]; __syncthreads(); }
    float var = red[0] * (1.0f / HN);
    out[(size_t)row * HN + t] = d * rsqrtf(var + 1e-5f) * g[t] + b[t];
}

// ---------- pooling / projection / logits ----------
__global__ __launch_bounds__(128) void pool_kernel(
    const float* __restrict__ hv, const int* __restrict__ bid,
    float* __restrict__ psum, float* __restrict__ pcnt)
{
    int row = blockIdx.x, t = threadIdx.x;
    int g = bid[row];
    atomicAdd(psum + (size_t)g * HN + t, hv[(size_t)row * HN + t]);
    if (t == 0) atomicAdd(pcnt + g, 1.0f);
}

__global__ __launch_bounds__(128) void proj_kernel(
    const float* __restrict__ psum, const float* __restrict__ pcnt,
    const float* __restrict__ P1, const float* __restrict__ P2,
    const float* __restrict__ pb, float* __restrict__ out)
{
    __shared__ float ge[128];
    __shared__ float t1[128];
    int g = blockIdx.x, t = threadIdx.x;
    float cnt = pcnt[g]; cnt = cnt < 1.f ? 1.f : cnt;
    ge[t] = psum[(size_t)g * HN + t] / cnt;
    __syncthreads();
    float a = 0.f;
    for (int k = 0; k < HN; ++k) a += ge[k] * P1[k * HN + t];
    t1[t] = fmaxf(a, 0.f);
    __syncthreads();
    float b = 0.f;
    for (int k = 0; k < HN; ++k) b += t1[k] * P2[k * HN + t];
    out[(size_t)g * HN + t] = b + pb[t];
}

__global__ void logit_kernel(const float* __restrict__ hv, const float* __restrict__ Wo,
                             const float* __restrict__ bo, float* __restrict__ out)
{
    int idx = blockIdx.x * blockDim.x + threadIdx.x;
    int n = idx >> 2, v = idx & 3;
    const float* hr = hv + (size_t)n * HN;
    float a = bo[v];
    for (int k = 0; k < HN; ++k) a += hr[k] * Wo[k * VV + v];
    out[idx] = a;
}

// ---------- launch ----------
extern "C" void kernel_launch(void* const* d_in, const int* in_sizes, int n_in,
                              void* d_out, int out_size, void* d_ws, size_t ws_size,
                              hipStream_t stream)
{
    (void)in_sizes; (void)n_in; (void)out_size; (void)ws_size;

    const float* hV0  = (const float*)d_in[0];
    const float* hE   = (const float*)d_in[1];
    const int*   Eidx = (const int*)d_in[2];
    const int*   bid  = (const int*)d_in[3];
    const int*   S    = (const int*)d_in[4];
    const float* Lw[6][14];
    for (int l = 0; l < 6; ++l)
        for (int k = 0; k < 14; ++k)
            Lw[l][k] = (const float*)d_in[5 + l * 14 + k];
    const float* P1 = (const float*)d_in[89];
    const float* P2 = (const float*)d_in[90];
    const float* pb = (const float*)d_in[91];
    const float* Wo = (const float*)d_in[92];
    const float* bo = (const float*)d_in[93];

    char* ws = (char*)d_ws;
    size_t off = 0;
    auto alloc = [&](size_t bytes) { void* p = ws + off; off = (off + bytes + 255) & ~(size_t)255; return p; };
    float*  hv   = (float*) alloc((size_t)NN * HN * 4);
    float*  dh   = (float*) alloc((size_t)NN * HN * 4);
    float*  ffb  = (float*) alloc((size_t)NN * HN * 4);
    __bf16* hvb  = (__bf16*)alloc((size_t)NN * HN * 2);
    float*  psum = (float*) alloc((size_t)BB * HN * 4);
    float*  pcnt = (float*) alloc((size_t)BB * 4);

    const int wk[5] = {384, 128, 128, 128, 512};
    const int wn[5] = {128, 128, 128, 512, 128};
    const int widx[5] = {0, 2, 4, 8, 10};  // W1,W2,W3,Wd1,Wd2
    __bf16* wsw[6][5];
    for (int l = 0; l < 6; ++l)
        for (int m = 0; m < 5; ++m)
            wsw[l][m] = (__bf16*)alloc((size_t)wk[m] * wn[m] * 2);

    // one-time weight swizzle (f32 -> fragment-layout bf16)
    for (int l = 0; l < 6; ++l)
        for (int m = 0; m < 5; ++m) {
            int el = wk[m] * wn[m];
            swz_kernel<<<el / 256, 256, 0, stream>>>(Lw[l][widx[m]], wsw[l][m], wk[m], wn[m]);
        }

    copyf_kernel<<<(NN * HN) / 256, 256, 0, stream>>>(hv, hV0, NN * HN);

    for (int l = 0; l < 6; ++l) {
        cvtbf_kernel<<<(NN * HN) / 256, 256, 0, stream>>>(hvb, hv, NN * HN);
        zerof_kernel<<<(NN * HN) / 256, 256, 0, stream>>>(dh, NN * HN);
        edge_kernel<<<(int)(EE / 64), 128, 0, stream>>>(
            hE, hvb, Eidx, (long)EE,
            wsw[l][0], wsw[l][1], wsw[l][2],
            Lw[l][1], Lw[l][3], Lw[l][5], dh);
        ln_kernel<<<NN, 128, 0, stream>>>(hv, dh, Lw[l][6], Lw[l][7], hv);
        ff_kernel<<<NN / 64, 128, 0, stream>>>(hv, wsw[l][3], wsw[l][4], Lw[l][9], Lw[l][11], ffb);
        ln_kernel<<<NN, 128, 0, stream>>>(hv, ffb, Lw[l][12], Lw[l][13], hv);
    }

    // pooling + projection + logits + S passthrough
    zerof_kernel<<<(BB * HN) / 256, 256, 0, stream>>>(psum, BB * HN);
    zerof_kernel<<<1, BB, 0, stream>>>(pcnt, BB);
    pool_kernel<<<NN, 128, 0, stream>>>(hv, bid, psum, pcnt);

    float* outF = (float*)d_out;
    logit_kernel<<<(NN * VV) / 256, 256, 0, stream>>>(hv, Wo, bo, outF);
    copyi_kernel<<<NN / 256, 256, 0, stream>>>((int*)(outF + NN * VV), S, NN);
    proj_kernel<<<BB, 128, 0, stream>>>(psum, pcnt, P1, P2, pb, outF + NN * VV + NN);
}
